// MultiHeadAttention_4930622456461
// MI455X (gfx1250) — matrix operand order repaired and checked by reading, not compiled, not run on hardware
//
#include <hip/hip_runtime.h>
#include <hip/hip_bf16.h>

// ---------------------------------------------------------------------------
// MHA block for MI455X (gfx1250, wave32, WMMA, TDM).
//   prep: x -> bf16, W_qkv/W_proj -> transposed bf16 (all B-operands become
//         K-contiguous 32B loads; whole working set is L2-resident: 192 MB)
//   qkv : xb @ Wqkv^T + b  -> Q,K [b][h][z][e] bf16, V stored transposed
//         Vt [b][h][e][z] so P@V B-fragments are contiguous
//   attn: Q slab staged to LDS via tensor_load_to_lds (TDM) ->
//         S = Q K^T (wmma) -> softmax/sqrt(e) * exp(-(j-i)^2/(z/2)) -> P@V
//   proj: O @ Wproj^T + b -> f32 out
// Register blocking: GEMM waves own 32x64 tiles (2 A x 4 B fragments ->
// 8 wmma per 12 b128 loads); attention shares one LDS A across 2/6 wmma.
// Wave-uniform tile coords go through readfirstlane so epilogues are scalar.
// ---------------------------------------------------------------------------

typedef __attribute__((ext_vector_type(16))) __bf16       v16bf;
typedef __attribute__((ext_vector_type(8)))  float        v8f;
typedef __attribute__((ext_vector_type(4)))  unsigned int v4u;
typedef __attribute__((ext_vector_type(8)))  int          v8i;
typedef __attribute__((ext_vector_type(4)))  int          v4i;

#define EMB   768
#define BB    8
#define NN    2048
#define HH    8
#define ZZ    256            // NN / HH
#define MTOT  (BB * NN)      // 16384 rows of x / O
#define MT2   (MTOT / 32)    // 512 32-row M-strips
#define DT4   (EMB / 64)     // 12 64-wide N-strips over emb dim

__device__ __forceinline__ v8f wmma_bf16(v16bf a, v16bf b, v8f c) {
  // (neg_a, A, neg_b, B, c_mod, C, reuse_a, reuse_b)
  return __builtin_amdgcn_wmma_f32_16x16x32_bf16(false, a, false, b, (short)0, c,
                                                 false, false);
}

// A fragment (16x32 bf16) from a row-major bf16 matrix, leading dim ld.
// Layout: lane m = L&15; lanes 0-15 hold K = k0..k0+15,
// lanes 16-31 hold K = k0+16..k0+31.  One contiguous 32B run.
__device__ __forceinline__ v16bf a_frag(const __bf16* A, int ld, int m0, int k0,
                                        int lane) {
  const int half = lane >> 4;
  const __bf16* row = A + (size_t)(m0 + (lane & 15)) * ld;
  v16bf r;
#pragma unroll
  for (int i = 0; i < 16; ++i) r[i] = row[k0 + 16 * half + i];
  return r;
}

// B fragment (32x16) where the WMMA K dim is CONTIGUOUS in memory; the caller
// supplies this lane's row base (row n of B^T).  lanes 0-15: K=k0..k0+15,
// lanes 16-31: K=k0+16..k0+31 -> one 32B contiguous load per lane.
__device__ __forceinline__ v16bf b_frag_row(const __bf16* row, int k0, int lane) {
  const __bf16* p = row + k0 + 16 * (lane >> 4);
  v16bf r;
#pragma unroll
  for (int i = 0; i < 16; ++i) r[i] = p[i];
  return r;
}

// ---------------------------------------------------------------------------
// TDM: stage a 2D bf16 tile (rows x cols, row stride in elements) into LDS.
// D# per CDNA5 ISA 8.3/8.4: group0 = {count=1, lds_addr, global_addr[56:0],
// type=2}; group1 = {mask=0, data_size=2B, dims/strides}.
// ---------------------------------------------------------------------------
#if defined(__has_builtin)
#if __has_builtin(__builtin_amdgcn_tensor_load_to_lds) && \
    __has_builtin(__builtin_amdgcn_s_wait_tensorcnt)
#define HAVE_TDM 1
#endif
#endif
#ifndef HAVE_TDM
#define HAVE_TDM 0
#endif

#if HAVE_TDM
__device__ __forceinline__ void tdm_load_2d_bf16(unsigned lds_off,
                                                 const void* gptr,
                                                 unsigned rows, unsigned cols,
                                                 unsigned row_stride) {
  const unsigned long long ga = (unsigned long long)(uintptr_t)gptr;
  v4u g0;
  g0[0] = 1u;                                       // count=1, user desc
  g0[1] = lds_off;                                  // lds_addr (bytes)
  g0[2] = (unsigned)(ga & 0xffffffffu);             // global_addr[31:0]
  g0[3] = (unsigned)((ga >> 32) & 0x1ffffffu) | (2u << 30);  // [56:32]|type=2
  v8i g1;
  g1[0] = 1 << 16;                                  // wg_mask=0, data_size=2B
  g1[1] = (int)((cols & 0xffffu) << 16);            // tensor_dim0[15:0]
  g1[2] = (int)(((cols >> 16) & 0xffffu) | ((rows & 0xffffu) << 16));
  g1[3] = (int)(((rows >> 16) & 0xffffu) | ((cols & 0xffffu) << 16)); // tile_dim0
  g1[4] = (int)(rows & 0xffffu);                    // tile_dim1 (tile_dim2=0)
  g1[5] = (int)row_stride;                          // tensor_dim0_stride[31:0]
  g1[6] = 0;
  g1[7] = 0;
  v4i gz = {0, 0, 0, 0};
#if __has_include(<hip/amd_detail/amd_gfx1250_TDM.h>)
  v8i gz8 = {0, 0, 0, 0, 0, 0, 0, 0};               // clang-23 6-arg form
  __builtin_amdgcn_tensor_load_to_lds(g0, g1, gz, gz, gz8, 0);
#else
  __builtin_amdgcn_tensor_load_to_lds(g0, g1, gz, gz, 0);  // ROCm 7.2 5-arg
#endif
}
#endif

// ---------------------------------------------------------------------------
// Kernel 0: one-shot precision/layout prep (everything stays L2-resident).
// ---------------------------------------------------------------------------
__global__ __launch_bounds__(256) void prep_kernel(
    const float* __restrict__ x, const float* __restrict__ Wqkv,
    const float* __restrict__ Wproj, __bf16* __restrict__ xb,
    __bf16* __restrict__ Wqkvt, __bf16* __restrict__ Wprojt) {
  const size_t stride = (size_t)gridDim.x * blockDim.x;
  const size_t tid = (size_t)blockIdx.x * blockDim.x + threadIdx.x;
  for (size_t i = tid; i < (size_t)MTOT * EMB; i += stride)
    xb[i] = (__bf16)x[i];
  for (size_t i = tid; i < (size_t)3 * EMB * EMB; i += stride) {
    const size_t c = i / EMB, k = i % EMB;
    Wqkvt[i] = (__bf16)Wqkv[k * (3 * EMB) + c];
  }
  for (size_t i = tid; i < (size_t)EMB * EMB; i += stride) {
    const size_t c = i / EMB, k = i % EMB;
    Wprojt[i] = (__bf16)Wproj[k * EMB + c];
  }
}

// ---------------------------------------------------------------------------
// Kernel 1: qkv = xb @ Wqkv^T + b_qkv.  One wave per 32x64 tile:
// 2 A-frags x 4 B-frags -> 8 wmma per k-step at 1.5 b128-load/wmma.
// Q,K -> [b][h][z][e];  V -> Vt [b][h][e][z].
// ---------------------------------------------------------------------------
__global__ __launch_bounds__(256) void qkv_kernel(
    const __bf16* __restrict__ xb, const __bf16* __restrict__ Wqkvt,
    const float* __restrict__ bqkv, __bf16* __restrict__ Q,
    __bf16* __restrict__ K, __bf16* __restrict__ Vt) {
  const int lane = threadIdx.x & 31;
  // Wave-uniform tile coordinates -> SGPRs (scalar epilogue branch).
  const int gw = __builtin_amdgcn_readfirstlane(blockIdx.x * 8 +
                                                (threadIdx.x >> 5));
  const int s = gw / (MT2 * DT4);     // 0=q 1=k 2=v
  if (s >= 3) return;
  const int rem = gw % (MT2 * DT4);
  const int m0 = (rem % MT2) * 32;
  const int d0 = (rem / MT2) * 64;
  const int nlane = lane & 15;

  const __bf16* arow = xb + (size_t)(m0 + nlane) * EMB;
  const __bf16* brow[4];
#pragma unroll
  for (int j = 0; j < 4; ++j)
    brow[j] = Wqkvt + (size_t)(3 * (d0 + 16 * j + nlane) + s) * EMB;

  v8f acc[2][4] = {{{}, {}, {}, {}}, {{}, {}, {}, {}}};
  for (int k0 = 0; k0 < EMB; k0 += 32) {
    __builtin_prefetch(arow + k0 + 64, 0, 0);       // global_prefetch_b8
    v16bf a0 = a_frag(xb, EMB, m0, k0, lane);
    v16bf a1 = a_frag(xb, EMB, m0 + 16, k0, lane);
#pragma unroll
    for (int j = 0; j < 4; ++j) {
      v16bf b = b_frag_row(brow[j], k0, lane);
      acc[0][j] = wmma_bf16(a0, b, acc[0][j]);
      acc[1][j] = wmma_bf16(a1, b, acc[1][j]);
    }
  }

  const int half = lane >> 4;
#pragma unroll
  for (int i = 0; i < 2; ++i) {
#pragma unroll
    for (int j = 0; j < 4; ++j) {
      const int dcol = d0 + 16 * j + nlane;
      const float bias = bqkv[3 * dcol + s];
#pragma unroll
      for (int r = 0; r < 8; ++r) {
        const int m = m0 + 16 * i + r + 8 * half;   // flat (b, token) row
        const int bidx = m / NN, nidx = m % NN;
        const int hi = nidx & 7, zi = nidx >> 3;    // token = zi*8 + hi
        const __bf16 val = (__bf16)(acc[i][j][r] + bias);
        if (s == 2)  // V transposed: [b][h][e][z]
          Vt[(((size_t)bidx * HH + hi) * EMB + dcol) * ZZ + zi] = val;
        else         // Q / K: [b][h][z][e]
          ((s == 0) ? Q : K)[(((size_t)bidx * HH + hi) * ZZ + zi) * EMB +
                             dcol] = val;
      }
    }
  }
}

// ---------------------------------------------------------------------------
// Kernel 2: attention. One block per (b, h, 16-query-row slab).
// ---------------------------------------------------------------------------
__global__ __launch_bounds__(256) void attn_kernel(
    const __bf16* __restrict__ Q, const __bf16* __restrict__ K,
    const __bf16* __restrict__ Vt, __bf16* __restrict__ O) {
  __shared__ __bf16 Qs[16][EMB];     // Q slab staged by the TDM (24 KB)
  __shared__ float  S[16][ZZ + 1];   // energy slab (f32)
  __shared__ __bf16 P[16][ZZ];       // softmax probs (A-operand for P@V)

  const int qb = blockIdx.x & 15;    // query row-block within head
  const int bh = blockIdx.x >> 4;    // flattened [b][h]
  const __bf16* Qp = Q + (size_t)bh * ZZ * EMB;
  const __bf16* Kp = K + (size_t)bh * ZZ * EMB;
  const __bf16* Vp = Vt + (size_t)bh * EMB * ZZ;
  const int wave = __builtin_amdgcn_readfirstlane(threadIdx.x >> 5);
  const int lane = threadIdx.x & 31;
  const int nlane = lane & 15, half = lane >> 4;
  const int m0 = qb * 16;

  // --- Stage Q slab (rows m0..m0+15, contiguous 16x768 bf16) into LDS -----
#if HAVE_TDM
  if (wave == 0) {
    tdm_load_2d_bf16((unsigned)(uintptr_t)&Qs[0][0], Qp + (size_t)m0 * EMB,
                     16u, (unsigned)EMB, (unsigned)EMB);
    __builtin_amdgcn_s_wait_tensorcnt((short)0);
  }
#else
  {
    const __bf16* src = Qp + (size_t)m0 * EMB;
    __bf16* dst = &Qs[0][0];
    for (int i = threadIdx.x; i < 16 * EMB; i += 256) dst[i] = src[i];
  }
#endif
  __syncthreads();

  // --- Phase 1: S = Q_slab @ K^T; each wave owns a 32-key strip (2 acc) ---
  {
    const int n0 = wave * 32;
    const __bf16* br0 = Kp + (size_t)(n0 + nlane) * EMB;       // K^T rows
    const __bf16* br1 = Kp + (size_t)(n0 + 16 + nlane) * EMB;
    v8f acc0 = {}, acc1 = {};
    for (int k0 = 0; k0 < EMB; k0 += 32) {
      v16bf a = a_frag(&Qs[0][0], EMB, 0, k0, lane);           // ds loads
      acc0 = wmma_bf16(a, b_frag_row(br0, k0, lane), acc0);
      acc1 = wmma_bf16(a, b_frag_row(br1, k0, lane), acc1);
    }
#pragma unroll
    for (int r = 0; r < 8; ++r) {
      S[r + 8 * half][n0 + nlane]      = acc0[r];
      S[r + 8 * half][n0 + 16 + nlane] = acc1[r];
    }
  }
  __syncthreads();

  // --- softmax / sqrt(e) * Gaussian decay: 16 lanes per row ---------------
  const int row = threadIdx.x >> 4, sub = threadIdx.x & 15;
  float mx = -1e30f;
  for (int j = sub; j < ZZ; j += 16) mx = fmaxf(mx, S[row][j]);
#pragma unroll
  for (int o = 8; o > 0; o >>= 1) mx = fmaxf(mx, __shfl_xor(mx, o, 16));
  float sum = 0.f;
  for (int j = sub; j < ZZ; j += 16) {
    const float e = __expf(S[row][j] - mx);
    S[row][j] = e;
    sum += e;
  }
#pragma unroll
  for (int o = 8; o > 0; o >>= 1) sum += __shfl_xor(sum, o, 16);
  const float inv = 1.0f / (sum * 27.712812921102035f);  // * sqrt(768)
  const int qi = m0 + row;                               // query z-index
  for (int j = sub; j < ZZ; j += 16) {
    const float d = (float)(j - qi);
    const float corr = __expf(-(d * d) * (1.0f / 128.0f));  // z/2 = 128
    P[row][j] = (__bf16)(S[row][j] * inv * corr);
  }
  __syncthreads();

  // --- Phase 2: O = P @ V; each wave owns a 96-wide emb strip (6 acc) -----
  {
    const int n0 = wave * 96;
    const int bidx = bh >> 3, hi = bh & 7;
    const __bf16* brow[6];
#pragma unroll
    for (int j = 0; j < 6; ++j)
      brow[j] = Vp + (size_t)(n0 + 16 * j + nlane) * ZZ;  // Vt rows
    v8f acc[6] = {{}, {}, {}, {}, {}, {}};
    for (int k0 = 0; k0 < ZZ; k0 += 32) {
      v16bf a = a_frag(&P[0][0], ZZ, 0, k0, lane);        // ds loads
#pragma unroll
      for (int j = 0; j < 6; ++j)
        acc[j] = wmma_bf16(a, b_frag_row(brow[j], k0, lane), acc[j]);
    }
#pragma unroll
    for (int j = 0; j < 6; ++j) {
      const int dcol = n0 + 16 * j + nlane;
#pragma unroll
      for (int r = 0; r < 8; ++r) {
        const int zi = m0 + r + 8 * half;
        O[((size_t)bidx * NN + (size_t)zi * HH + hi) * EMB + dcol] =
            (__bf16)acc[j][r];
      }
    }
  }
}

// ---------------------------------------------------------------------------
// Kernel 3: out = O @ Wproj^T + b_proj.  One wave per 32x64 tile.
// ---------------------------------------------------------------------------
__global__ __launch_bounds__(256) void proj_kernel(
    const __bf16* __restrict__ O, const __bf16* __restrict__ Wprojt,
    const float* __restrict__ bp, float* __restrict__ out) {
  const int lane = threadIdx.x & 31;
  const int gw = __builtin_amdgcn_readfirstlane(blockIdx.x * 8 +
                                                (threadIdx.x >> 5));
  if (gw >= MT2 * DT4) return;             // wave-uniform
  const int m0 = (gw % MT2) * 32;
  const int n0 = (gw / MT2) * 64;
  const int nlane = lane & 15, half = lane >> 4;
  const __bf16* arow = O + (size_t)(m0 + nlane) * EMB;
  const __bf16* brow[4];
#pragma unroll
  for (int j = 0; j < 4; ++j)
    brow[j] = Wprojt + (size_t)(n0 + 16 * j + nlane) * EMB;

  v8f acc[2][4] = {{{}, {}, {}, {}}, {{}, {}, {}, {}}};
  for (int k0 = 0; k0 < EMB; k0 += 32) {
    __builtin_prefetch(arow + k0 + 64, 0, 0);
    v16bf a0 = a_frag(O, EMB, m0, k0, lane);
    v16bf a1 = a_frag(O, EMB, m0 + 16, k0, lane);
#pragma unroll
    for (int j = 0; j < 4; ++j) {
      v16bf b = b_frag_row(brow[j], k0, lane);
      acc[0][j] = wmma_bf16(a0, b, acc[0][j]);
      acc[1][j] = wmma_bf16(a1, b, acc[1][j]);
    }
  }
#pragma unroll
  for (int i = 0; i < 2; ++i) {
#pragma unroll
    for (int j = 0; j < 4; ++j) {
      const int c = n0 + 16 * j + nlane;
      const float bias = bp[c];
#pragma unroll
      for (int r = 0; r < 8; ++r)
        out[(size_t)(m0 + 16 * i + r + 8 * half) * EMB + c] =
            acc[i][j][r] + bias;
    }
  }
}

// ---------------------------------------------------------------------------
extern "C" void kernel_launch(void* const* d_in, const int* in_sizes, int n_in,
                              void* d_out, int out_size, void* d_ws,
                              size_t ws_size, hipStream_t stream) {
  (void)in_sizes; (void)n_in; (void)out_size; (void)ws_size;
  const float* x     = (const float*)d_in[0];
  const float* Wqkv  = (const float*)d_in[1];
  const float* bqkv  = (const float*)d_in[2];
  const float* Wproj = (const float*)d_in[3];
  const float* bproj = (const float*)d_in[4];
  float* out = (float*)d_out;

  // Workspace (bf16): xb + Wqkvt + Wprojt + Q + K + Vt + O  (~131 MB).
  const size_t e_x  = (size_t)MTOT * EMB;       // 12.58M
  const size_t e_wq = (size_t)3 * EMB * EMB;    //  1.77M
  const size_t e_wp = (size_t)EMB * EMB;        //  0.59M
  __bf16* xb     = (__bf16*)d_ws;
  __bf16* Wqkvt  = xb + e_x;
  __bf16* Wprojt = Wqkvt + e_wq;
  __bf16* Q      = Wprojt + e_wp;
  __bf16* K      = Q + e_x;
  __bf16* Vt     = K + e_x;
  __bf16* O      = Vt + e_x;

  prep_kernel<<<dim3(4096),               dim3(256), 0, stream>>>(
      x, Wqkv, Wproj, xb, Wqkvt, Wprojt);
  qkv_kernel <<<dim3((MT2 * DT4 * 3) / 8), dim3(256), 0, stream>>>(
      xb, Wqkvt, bqkv, Q, K, Vt);
  attn_kernel<<<dim3(BB * HH * 16),       dim3(256), 0, stream>>>(
      Q, K, Vt, O);
  proj_kernel<<<dim3((MT2 * DT4) / 8),    dim3(256), 0, stream>>>(
      O, Wprojt, bproj, out);
}
